// _PointnetSAModuleBase_74577812128433
// MI455X (gfx1250) — matrix operand order repaired and checked by reading, not compiled, not run on hardware
//
#include <hip/hip_runtime.h>

typedef __attribute__((ext_vector_type(16))) _Float16 v16h;
typedef __attribute__((ext_vector_type(8)))  float    v8f;

#define BB    4
#define NN    8192
#define CC    128
#define SS    2048
#define NSAMP 32
#define C1    64
#define C2    64
#define C3    128
#define K1    131
#define K1P   160          // permuted/padded K for layer1: [x,y,z,pad, feat0..127, pad..159]
#define RAD2  0.64f
#define WAVES 2

#define W1H_OFF 0
#define W2H_OFF (C1 * K1P)                    // 10240
#define W3H_OFF (C1 * K1P + C2 * C1)          // 14336
#define WH_TOTAL (C1 * K1P + C2 * C1 + C3 * C2)   // 22528 halves

// ---------------- Kernel 0: one-time f32 -> f16 weight repack (K-permuted W1) -----------
__global__ void __launch_bounds__(256) prep_weights_kernel(
    const float* __restrict__ W1, const float* __restrict__ W2,
    const float* __restrict__ W3, _Float16* __restrict__ wh)
{
    int i = blockIdx.x * blockDim.x + threadIdx.x;
    if (i < C1 * K1P) {
        int o = i / K1P, kp = i % K1P;
        float v = 0.f;
        if (kp < 3)                      v = W1[o * K1 + kp];
        else if (kp >= 4 && kp < 4 + CC) v = W1[o * K1 + kp - 1];
        wh[i] = (_Float16)v;
    } else if (i < C1 * K1P + C2 * C1) {
        wh[i] = (_Float16)W2[i - C1 * K1P];
    } else if (i < WH_TOTAL) {
        wh[i] = (_Float16)W3[i - C1 * K1P - C2 * C1];
    }
}

// ---------------- Kernel 1: ball query (first-32-in-index-order within radius) ----------
__global__ void __launch_bounds__(128) ball_query_kernel(
    const float* __restrict__ xyz, int* __restrict__ gidx,
    float* __restrict__ out_newxyz, float* __restrict__ out_idx)
{
    int wave = threadIdx.x >> 5;
    int lane = threadIdx.x & 31;
    int cid  = blockIdx.x * (blockDim.x >> 5) + wave;
    if (cid >= BB * SS) return;
    int b = cid / SS, s = cid % SS;
    int ci = (s < SS / 2) ? s : (s - SS / 2 + NN / 2);
    const float* xb = xyz + (size_t)b * NN * 3;
    float cx = xb[ci * 3 + 0], cy = xb[ci * 3 + 1], cz = xb[ci * 3 + 2];
    int base = cid * NSAMP;
    int cnt = 0, first = -1;
    for (int n0 = 0; n0 < NN; n0 += 32) {
        int p = n0 + lane;
        float dx = xb[p * 3 + 0] - cx;
        float dy = xb[p * 3 + 1] - cy;
        float dz = xb[p * 3 + 2] - cz;
        bool within = (dx * dx + dy * dy + dz * dz) < RAD2;
        unsigned m = (unsigned)__ballot(within);
        if (first < 0 && m != 0u) first = n0 + __builtin_ctz(m);
        int slot = cnt + __builtin_popcount(m & ((1u << lane) - 1u));
        if (within && slot < NSAMP) gidx[base + slot] = p;
        cnt += __builtin_popcount(m);
        if (cnt >= NSAMP) break;
    }
    if (first < 0) first = 0;
    if (cnt > NSAMP) cnt = NSAMP;
    if (lane >= cnt) gidx[base + lane] = first;   // fill empty slots with first hit
    if (lane == 0) {
        out_newxyz[cid * 3 + 0] = cx;
        out_newxyz[cid * 3 + 1] = cy;
        out_newxyz[cid * 3 + 2] = cz;
        out_idx[cid] = (float)ci;
    }
}

// ---------------- fragment helpers (CDNA5 wave32 WMMA operands, contiguous k = 16*hi + s) ---------------
__device__ __forceinline__ unsigned pack2h(float a, float b) {
    union { _Float16 h[2]; unsigned u; } t;
    t.h[0] = (_Float16)a; t.h[1] = (_Float16)b;
    return t.u;
}

// A-matrix 16x32 f16 from LDS: lane(lo,hi), row M=lo.
// Halves = 16 consecutive K starting at kbase+16*hi  (two 16B chunks)
__device__ __forceinline__ v16h load_a_frag(const _Float16* rowbase, int kbase, int hi) {
    union { v16h h; uint4 q[2]; } f;
    f.q[0] = *(const uint4*)(rowbase + kbase + hi * 16);
    f.q[1] = *(const uint4*)(rowbase + kbase + hi * 16 + 8);
    return f.h;
}

// B-matrix 32x16 f16 from pre-packed global f16 weights:
// lane(lo,hi) -> N=lo, halves = 16 consecutive K starting at 32*kt + 16*hi  (32 contiguous B)
__device__ __forceinline__ v16h load_b_frag(const _Float16* __restrict__ p) {
    union { v16h h; uint4 q[2]; } f;
    f.q[0] = *(const uint4*)(p);
    f.q[1] = *(const uint4*)(p + 8);
    return f.h;
}

// ---------------- Kernel 2: grouped MLP via WMMA, one wave per center -------------------
__global__ void __launch_bounds__(WAVES * 32) sa_mlp_kernel(
    const float* __restrict__ xyz,  const float* __restrict__ feat,
    const _Float16* __restrict__ wh,
    const float* __restrict__ b1,   const float* __restrict__ b2,
    const float* __restrict__ b3,
    const int*   __restrict__ gidx, float* __restrict__ out_feat)
{
    __shared__ __align__(16) _Float16 Gs [WAVES][32][K1P];
    __shared__ __align__(16) _Float16 H1s[WAVES][32][C1];
    __shared__ __align__(16) _Float16 H2s[WAVES][32][C2];

    const _Float16* __restrict__ W1h = wh + W1H_OFF;
    const _Float16* __restrict__ W2h = wh + W2H_OFF;
    const _Float16* __restrict__ W3h = wh + W3H_OFF;

    int w    = threadIdx.x >> 5;
    int lane = threadIdx.x & 31;
    int lo   = lane & 15;
    int hi   = lane >> 4;
    int cid  = blockIdx.x * WAVES + w;
    if (cid >= BB * SS) return;
    int b = cid / SS, s = cid % SS;
    int ci = (s < SS / 2) ? s : (s - SS / 2 + NN / 2);

    const float* xb = xyz + (size_t)b * NN * 3;
    float cx = xb[ci * 3 + 0], cy = xb[ci * 3 + 1], cz = xb[ci * 3 + 2];

    // ---- stage G (32 x 160 f16) : lane j builds row j (b64 LDS stores) ----
    int g = gidx[cid * NSAMP + lane];
    _Float16* grow = &Gs[w][lane][0];
    float px = xb[g * 3 + 0] - cx;
    float py = xb[g * 3 + 1] - cy;
    float pz = xb[g * 3 + 2] - cz;
    *(uint2*)(grow) = make_uint2(pack2h(px, py), pack2h(pz, 0.f));
    const float4* frow = (const float4*)(feat + ((size_t)b * NN + g) * CC);
#pragma unroll
    for (int c4 = 0; c4 < CC / 4; ++c4) {
        float4 f4 = frow[c4];
        *(uint2*)(grow + 4 + c4 * 4) =
            make_uint2(pack2h(f4.x, f4.y), pack2h(f4.z, f4.w));
    }
#pragma unroll
    for (int c = 4 + CC; c < K1P; c += 4)
        *(uint2*)(grow + c) = make_uint2(0u, 0u);
    asm volatile("" ::: "memory");   // DS ops are in-order within a wave; fence the compiler

    const v8f vzero = {0.f, 0.f, 0.f, 0.f, 0.f, 0.f, 0.f, 0.f};

    // ---- layer 1: H1 = relu(G @ W1^T + b1)   (M=32, K=160(pad), N=64) ----
    v8f acc1[2][4];
#pragma unroll
    for (int mt = 0; mt < 2; ++mt)
#pragma unroll
        for (int nt = 0; nt < 4; ++nt) acc1[mt][nt] = vzero;

#pragma unroll
    for (int kt = 0; kt < K1P / 32; ++kt) {
        v16h a0 = load_a_frag(&Gs[w][lo][0],      kt * 32, hi);
        v16h a1 = load_a_frag(&Gs[w][16 + lo][0], kt * 32, hi);
#pragma unroll
        for (int nt = 0; nt < 4; ++nt) {
            v16h bf = load_b_frag(W1h + (nt * 16 + lo) * K1P + kt * 32 + hi * 16);
            acc1[0][nt] = __builtin_amdgcn_wmma_f32_16x16x32_f16(
                false, a0, false, bf, (short)0, acc1[0][nt], false, false);
            acc1[1][nt] = __builtin_amdgcn_wmma_f32_16x16x32_f16(
                false, a1, false, bf, (short)0, acc1[1][nt], false, false);
        }
    }
#pragma unroll
    for (int nt = 0; nt < 4; ++nt) {
        float bias = b1[nt * 16 + lo];
#pragma unroll
        for (int mt = 0; mt < 2; ++mt)
#pragma unroll
            for (int r = 0; r < 8; ++r) {
                float v = acc1[mt][nt][r] + bias;
                v = v > 0.f ? v : 0.f;
                H1s[w][mt * 16 + hi * 8 + r][nt * 16 + lo] = (_Float16)v;
            }
    }
    asm volatile("" ::: "memory");

    // ---- layer 2: H2 = relu(H1 @ W2^T + b2)   (K=64, N=64) ----
    v8f acc2[2][4];
#pragma unroll
    for (int mt = 0; mt < 2; ++mt)
#pragma unroll
        for (int nt = 0; nt < 4; ++nt) acc2[mt][nt] = vzero;
#pragma unroll
    for (int kt = 0; kt < C1 / 32; ++kt) {
        v16h a0 = load_a_frag(&H1s[w][lo][0],      kt * 32, hi);
        v16h a1 = load_a_frag(&H1s[w][16 + lo][0], kt * 32, hi);
#pragma unroll
        for (int nt = 0; nt < 4; ++nt) {
            v16h bf = load_b_frag(W2h + (nt * 16 + lo) * C1 + kt * 32 + hi * 16);
            acc2[0][nt] = __builtin_amdgcn_wmma_f32_16x16x32_f16(
                false, a0, false, bf, (short)0, acc2[0][nt], false, false);
            acc2[1][nt] = __builtin_amdgcn_wmma_f32_16x16x32_f16(
                false, a1, false, bf, (short)0, acc2[1][nt], false, false);
        }
    }
#pragma unroll
    for (int nt = 0; nt < 4; ++nt) {
        float bias = b2[nt * 16 + lo];
#pragma unroll
        for (int mt = 0; mt < 2; ++mt)
#pragma unroll
            for (int r = 0; r < 8; ++r) {
                float v = acc2[mt][nt][r] + bias;
                v = v > 0.f ? v : 0.f;
                H2s[w][mt * 16 + hi * 8 + r][nt * 16 + lo] = (_Float16)v;
            }
    }
    asm volatile("" ::: "memory");

    // ---- layer 3 + max over 32 samples: (K=64, N=128) ----
    // Hoist the 4 A-fragments (kt x mt) out of the 8-wide N loop.
    v16h a3[2][2];
#pragma unroll
    for (int kt = 0; kt < C2 / 32; ++kt) {
        a3[kt][0] = load_a_frag(&H2s[w][lo][0],      kt * 32, hi);
        a3[kt][1] = load_a_frag(&H2s[w][16 + lo][0], kt * 32, hi);
    }
    float* outb = out_feat + (size_t)b * C3 * SS + s;
#pragma unroll
    for (int nt = 0; nt < 8; ++nt) {
        v8f a0acc = vzero, a1acc = vzero;
#pragma unroll
        for (int kt = 0; kt < C2 / 32; ++kt) {
            v16h bf = load_b_frag(W3h + (nt * 16 + lo) * C2 + kt * 32 + hi * 16);
            a0acc = __builtin_amdgcn_wmma_f32_16x16x32_f16(
                false, a3[kt][0], false, bf, (short)0, a0acc, false, false);
            a1acc = __builtin_amdgcn_wmma_f32_16x16x32_f16(
                false, a3[kt][1], false, bf, (short)0, a1acc, false, false);
        }
        float bias = b3[nt * 16 + lo];
        float mx = 0.f;   // post-relu values are >= 0
#pragma unroll
        for (int r = 0; r < 8; ++r) {
            float v0 = a0acc[r] + bias; v0 = v0 > 0.f ? v0 : 0.f;
            float v1 = a1acc[r] + bias; v1 = v1 > 0.f ? v1 : 0.f;
            mx = fmaxf(mx, fmaxf(v0, v1));
        }
        // combine half-waves: lane lo and lane lo+16 hold the two M-halves for N=lo
        mx = fmaxf(mx, __shfl_xor(mx, 16, 32));
        if (hi == 0) outb[(size_t)(nt * 16 + lo) * SS] = mx;
    }
}

// ---------------- host launch -----------------------------------------------------------
extern "C" void kernel_launch(void* const* d_in, const int* in_sizes, int n_in,
                              void* d_out, int out_size, void* d_ws, size_t ws_size,
                              hipStream_t stream) {
    const float* xyz  = (const float*)d_in[0];
    const float* feat = (const float*)d_in[1];
    const float* W1   = (const float*)d_in[2];
    const float* b1   = (const float*)d_in[3];
    const float* W2   = (const float*)d_in[4];
    const float* b2   = (const float*)d_in[5];
    const float* W3   = (const float*)d_in[6];
    const float* b3   = (const float*)d_in[7];

    float* out        = (float*)d_out;
    float* out_newxyz = out;                                  // B*S*3
    float* out_feat   = out + (size_t)BB * SS * 3;            // B*C3*S
    float* out_idx    = out + (size_t)BB * SS * 3 + (size_t)BB * C3 * SS;  // B*S

    int*       gidx = (int*)d_ws;                                           // B*S*32 ints
    _Float16*  wh   = (_Float16*)((char*)d_ws + (size_t)BB * SS * NSAMP * 4);  // f16 weights

    prep_weights_kernel<<<(WH_TOTAL + 255) / 256, 256, 0, stream>>>(W1, W2, W3, wh);
    ball_query_kernel<<<(BB * SS) / 4, 128, 0, stream>>>(xyz, gidx, out_newxyz, out_idx);
    sa_mlp_kernel<<<(BB * SS) / WAVES, WAVES * 32, 0, stream>>>(
        xyz, feat, wh, b1, b2, b3, gidx, out_feat);
}
